// MutiScaleAttention_nhwc_28140625723665
// MI455X (gfx1250) — matrix operand order repaired and checked by reading, not compiled, not run on hardware
//
#include <hip/hip_runtime.h>
#include <hip/hip_bf16.h>
#include <stdint.h>

typedef unsigned short u16;
typedef __attribute__((ext_vector_type(16))) __bf16 v16bf;
typedef __attribute__((ext_vector_type(8)))  float  v8f;
typedef unsigned int u32x4 __attribute__((ext_vector_type(4)));
typedef unsigned int u32x8 __attribute__((ext_vector_type(8)));

union BF16Frag { v16bf v; uint4 q[2]; };

__device__ __forceinline__ u16 f2bf(float f){
  unsigned u = __float_as_uint(f);
  u += 0x7FFFu + ((u >> 16) & 1u);   // round-to-nearest-even
  return (u16)(u >> 16);
}
__device__ __forceinline__ float bf2f(u16 h){ return __uint_as_float(((unsigned)h) << 16); }

__device__ __forceinline__ v8f zero8(){
  v8f z;
#pragma unroll
  for(int i=0;i<8;++i) z[i]=0.f;
  return z;
}

// A-fragment (16x32 bf16): lane m=lane&15, kh=lane>>4.
// halves 0..7  -> K = kh*8 + 0..7 ; halves 8..15 -> K = 16 + kh*8 + 0..7
__device__ __forceinline__ v16bf load_frag_a(const u16* p, int kh){
  BF16Frag f;
  f.q[0] = *(const uint4*)(p + kh*8);
  f.q[1] = *(const uint4*)(p + kh*8 + 16);
  return f.v;
}
// B-fragment (32x16 bf16): lane n=lane&15, kg=lane>>4, halves 0..7 -> K = kg*8 + 0..7 ; halves 8..15 -> K = 16 + kg*8 + 0..7.
__device__ __forceinline__ v16bf load_frag_b(const u16* p, int kg){
  BF16Frag f;
  f.q[0] = *(const uint4*)(p + kg*8);
  f.q[1] = *(const uint4*)(p + kg*8 + 16);
  return f.v;
}

#define WMMA_BF16(A,B,C) __builtin_amdgcn_wmma_f32_16x16x32_bf16(false,(A),false,(B),(short)0,(C),false,false)

// Issue a 2D TDM tile load (global -> LDS). Descriptor per CDNA5 ISA §8.3/8.4.
// rows x cols elements of 2 bytes; optional LDS row padding via pad codes.
__device__ __forceinline__ void tdm_load_2d_b16(unsigned lds_addr, const void* gptr,
                                                unsigned cols, unsigned rows,
                                                unsigned row_stride_elems,
                                                int pad_enable, unsigned pad_interval_code,
                                                unsigned pad_amount_code)
{
  unsigned long long ga = (unsigned long long)gptr;
  u32x4 g0;
  g0.x = 1u;                                   // count=1, user descriptor
  g0.y = lds_addr;                             // LDS byte offset
  g0.z = (unsigned)(ga & 0xFFFFFFFFu);         // global_addr[31:0]
  g0.w = (unsigned)((ga >> 32) & 0x1FFFFFFu)   // global_addr[56:32]
       | (2u << 30);                           // type = 2 ("image")
  u32x8 g1;
  g1.s0 = (1u << 16)                                           // data_size = 2 bytes
        | ((unsigned)(pad_enable ? 1u : 0u) << 20)
        | (pad_interval_code << 22)
        | (pad_amount_code  << 25);
  g1.s1 = (cols & 0xFFFFu) << 16;              // tensor_dim0[15:0]
  g1.s2 = ((cols >> 16) & 0xFFFFu)             // tensor_dim0[31:16]
        | ((rows & 0xFFFFu) << 16);            // tensor_dim1[15:0]
  g1.s3 = ((rows >> 16) & 0xFFFFu)             // tensor_dim1[31:16]
        | ((cols & 0xFFFFu) << 16);            // tile_dim0
  g1.s4 = rows & 0xFFFFu;                      // tile_dim1 (tile_dim2 = 0)
  g1.s5 = row_stride_elems;                    // tensor_dim0_stride[31:0]
  g1.s6 = 0u;                                  // stride0[47:32] | dim1_stride[15:0]
  g1.s7 = 0u;
  asm volatile("tensor_load_to_lds %0, %1" :: "s"(g0), "s"(g1) : "memory");
}

// ---------------------------------------------------------------- utilities
__global__ void cvt_f32_bf16(const float* __restrict__ src, u16* __restrict__ dst, int n){
  for(int i = blockIdx.x*blockDim.x + threadIdx.x; i < n; i += gridDim.x*blockDim.x)
    dst[i] = f2bf(src[i]);
}
__global__ void zero_u16(u16* __restrict__ dst, int n){
  for(int i = blockIdx.x*blockDim.x + threadIdx.x; i < n; i += gridDim.x*blockDim.x)
    dst[i] = 0;
}

// ---------------------------------------------------------------- GEMM (K=384)
// C[m,n] = A[m,:384] * W[:384, n].  mode 0: split-store qkv (bf16, per-head),
// mode 1: fp32 out + bias.  Block tile 128x128, 8 waves of 64x32 (4x2 WMMA tiles).
__global__ __launch_bounds__(256) void gemm_bf16_k384(
    const u16* __restrict__ A, const u16* __restrict__ W, int Nout,
    int mode, const float* __restrict__ bias, float* __restrict__ outf,
    u16* __restrict__ outq, u16* __restrict__ outk, u16* __restrict__ outv)
{
  __shared__ __align__(16) u16 wT[128*392];   // [n][k], padded stride 392 halves
  const int tid = threadIdx.x, lane = tid & 31, wid = tid >> 5;
  const int wm = wid >> 2, wn = wid & 3;
  const int m15 = lane & 15, kh = lane >> 4;
  const int m0 = blockIdx.x * 128;
  const int n0 = blockIdx.y * 128;

  for(int idx = tid; idx < 128*384; idx += 256){
    int n = idx / 384, k = idx - n*384;
    wT[n*392 + k] = W[(size_t)k*Nout + n0 + n];
  }
  __syncthreads();

  v8f acc[4][2];
#pragma unroll
  for(int mi=0;mi<4;++mi)
#pragma unroll
    for(int ni=0;ni<2;++ni) acc[mi][ni] = zero8();

  const u16* arow[4];
#pragma unroll
  for(int mi=0;mi<4;++mi)
    arow[mi] = A + (size_t)(m0 + wm*64 + mi*16 + m15) * 384;

  for(int ks = 0; ks < 384; ks += 32){
    if(ks + 32 < 384) __builtin_prefetch(arow[0] + ks + 32, 0, 1);
    v16bf a[4], b[2];
#pragma unroll
    for(int mi=0;mi<4;++mi) a[mi] = load_frag_a(arow[mi] + ks, kh);
#pragma unroll
    for(int ni=0;ni<2;++ni) b[ni] = load_frag_b(&wT[(wn*32 + ni*16 + m15)*392 + ks], kh);
#pragma unroll
    for(int mi=0;mi<4;++mi)
#pragma unroll
      for(int ni=0;ni<2;++ni)
        acc[mi][ni] = WMMA_BF16(a[mi], b[ni], acc[mi][ni]);
  }

#pragma unroll
  for(int mi=0;mi<4;++mi){
#pragma unroll
    for(int ni=0;ni<2;++ni){
#pragma unroll
      for(int r=0;r<8;++r){
        int m = m0 + wm*64 + mi*16 + r + 8*kh;   // C-layout: VGPR r -> M=r (lanes 0-15), r+8 (16-31)
        int c = n0 + wn*32 + ni*16 + m15;
        float v = acc[mi][ni][r];
        if(mode == 1){
          outf[(size_t)m*Nout + c] = v + bias[c];
        } else {
          int b_ = m >> 12, tok = m & 4095;
          if(c < 384){      int h=c/48,      d=c-h*48;      outq[((size_t)(b_*8+h)*4096 + tok)*64 + d] = f2bf(v); }
          else if(c < 768){ int cc=c-384; int h=cc/48, d=cc-h*48; outk[((size_t)(b_*8+h)*4096 + tok)*48 + d] = f2bf(v); }
          else {            int cc=c-768; int h=cc/48, d=cc-h*48; outv[((size_t)(b_*8+h)*4096 + tok)*48 + d] = f2bf(v); }
        }
      }
    }
  }
}

// ---------------------------------------------------------------- pyramid pool
// src: [bh][4096][48] bf16. transpose==0 -> dst kp [bh][288][64] (l-major, padded)
// transpose==1 -> dst vpT [bh][48][288] (d-major).  L order: s=1 (row 0),
// s=2 (1..4), s=4 (5..20), s=16 (21..276); rows/cols >=277 and d>=48 zeroed.
__global__ __launch_bounds__(256) void psp_pool(const u16* __restrict__ src,
                                                u16* __restrict__ dst, int transpose)
{
  __shared__ float s16sum[256][48];   // raw sums over 4x4 pixel cells
  const int bh = blockIdx.x, t = threadIdx.x;
  const u16* sp = src + (size_t)bh*4096*48;

  const int cy = t >> 4, cx = t & 15;
  for(int d = 0; d < 48; ++d){
    float s = 0.f;
#pragma unroll
    for(int py=0; py<4; ++py)
#pragma unroll
      for(int px=0; px<4; ++px){
        int n = (cy*4 + py)*64 + (cx*4 + px);
        s += bf2f(sp[(size_t)n*48 + d]);
      }
    s16sum[t][d] = s;
  }
  __syncthreads();

  auto writeL = [&](int l, int d, float val){
    if(!transpose) dst[((size_t)bh*288 + l)*64 + d] = f2bf(val);
    else           dst[((size_t)bh*48  + d)*288 + l] = f2bf(val);
  };

  for(int d=0; d<48; ++d) writeL(21 + t, d, s16sum[t][d] * (1.f/16.f));   // s=16

  if(t < 16){                                                              // s=4
    int qy = t >> 2, qx = t & 3;
    for(int d=0; d<48; ++d){
      float s = 0.f;
      for(int yy=0; yy<4; ++yy) for(int xx=0; xx<4; ++xx)
        s += s16sum[(qy*4+yy)*16 + qx*4+xx][d];
      writeL(5 + t, d, s * (1.f/256.f));
    }
  } else if(t < 20){                                                       // s=2
    int i = t - 16, hy = i >> 1, hx = i & 1;
    for(int d=0; d<48; ++d){
      float s = 0.f;
      for(int yy=0; yy<8; ++yy) for(int xx=0; xx<8; ++xx)
        s += s16sum[(hy*8+yy)*16 + hx*8+xx][d];
      writeL(1 + i, d, s * (1.f/1024.f));
    }
  } else if(t == 20){                                                      // s=1
    for(int d=0; d<48; ++d){
      float s = 0.f;
      for(int c=0; c<256; ++c) s += s16sum[c][d];
      writeL(0, d, s * (1.f/4096.f));
    }
  }

  if(!transpose){
    for(int idx = t; idx < 288*64; idx += 256){
      int l = idx >> 6, d = idx & 63;
      if(l >= 277 || d >= 48) dst[((size_t)bh*288 + l)*64 + d] = 0;
    }
  } else {
    for(int idx = t; idx < 48*288; idx += 256){
      int d = idx / 288, l = idx - d*288;
      if(l >= 277) dst[((size_t)bh*48 + d)*288 + l] = 0;
    }
  }
}

// ---------------------------------------------------------------- attention
// One block = one (b,h) x 64 query rows; 4 waves x 16 rows.
// q: [bh][4096][64] (hd padded 48->64, zeros), kp: [bh][288][64], vpT: [bh][48][288].
// kp/vpT are staged into LDS by the Tensor Data Mover (wave 0 issues, TENSORcnt).
__global__ __launch_bounds__(128) void attn_psp(
    const u16* __restrict__ qbf, const u16* __restrict__ kpbf,
    const u16* __restrict__ vptbf, u16* __restrict__ aobf)
{
  __shared__ __align__(16) u16 kpL[288*72];        // TDM-padded stride 72 halves
  __shared__ __align__(16) u16 vpL[48*288];        // straight TDM copy, stride 288
  __shared__ __align__(16) u16 pbuf[4][16*296];    // per-wave normalized probs
  const int tid = threadIdx.x, lane = tid & 31, wid = tid >> 5;
  const int m15 = lane & 15, kh = lane >> 4;
  const int rb = blockIdx.x, bh = blockIdx.y;
  const float scale = 0.14433756729740643f;        // 48^-0.5

  if(tid < 32){
    // kp: 288 rows x 64 halves; LDS rows padded 64 -> 72 halves by TDM:
    // pad_interval code 4 => every 32 DWORDs (=one 128B row), pad_amount code 3 => 4 DWORDs.
    tdm_load_2d_b16((unsigned)(size_t)(void*)kpL,
                    kpbf + (size_t)bh*288*64, 64u, 288u, 64u,
                    1, 4u, 3u);
    // vpT: 48 rows x 288 halves, contiguous copy (no padding).
    tdm_load_2d_b16((unsigned)(size_t)(void*)vpL,
                    vptbf + (size_t)bh*48*288, 288u, 48u, 288u,
                    0, 0u, 0u);
    __builtin_amdgcn_s_wait_tensorcnt(0);
  }
  __syncthreads();

  const u16* qrow = qbf + ((size_t)bh*4096 + rb*64 + wid*16 + m15) * 64;
  v16bf aq0 = load_frag_a(qrow,      kh);
  v16bf aq1 = load_frag_a(qrow + 32, kh);

  // ---- S = Q Kp^T over 18 l-tiles (K = 64 via two 16x16x32 WMMAs)
  float sc[18][8];
#pragma unroll
  for(int t=0; t<18; ++t){
    v8f s = zero8();
    v16bf b0 = load_frag_b(&kpL[(t*16 + m15)*72],      kh);
    v16bf b1 = load_frag_b(&kpL[(t*16 + m15)*72 + 32], kh);
    s = WMMA_BF16(aq0, b0, s);
    s = WMMA_BF16(aq1, b1, s);
#pragma unroll
    for(int r=0;r<8;++r) sc[t][r] = s[r];
  }

  // ---- softmax over L (mask l>=277), reductions within each 16-lane half
  float mx[8];
#pragma unroll
  for(int r=0;r<8;++r) mx[r] = -3.0e38f;
#pragma unroll
  for(int t=0; t<18; ++t){
    bool valid = (t*16 + m15) < 277;
#pragma unroll
    for(int r=0;r<8;++r){
      float v = valid ? sc[t][r] : -3.0e38f;
      sc[t][r] = v;
      if(v > mx[r]) mx[r] = v;
    }
  }
#pragma unroll
  for(int off=1; off<16; off<<=1)
#pragma unroll
    for(int r=0;r<8;++r) mx[r] = fmaxf(mx[r], __shfl_xor(mx[r], off, 32));

  float sum[8];
#pragma unroll
  for(int r=0;r<8;++r) sum[r] = 0.f;
#pragma unroll
  for(int t=0; t<18; ++t)
#pragma unroll
    for(int r=0;r<8;++r){
      float e = __expf((sc[t][r] - mx[r]) * scale);
      sc[t][r] = e;
      sum[r] += e;
    }
#pragma unroll
  for(int off=1; off<16; off<<=1)
#pragma unroll
    for(int r=0;r<8;++r) sum[r] += __shfl_xor(sum[r], off, 32);

  float inv[8];
#pragma unroll
  for(int r=0;r<8;++r) inv[r] = 1.f / sum[r];

  u16* pb = pbuf[wid];
#pragma unroll
  for(int t=0; t<18; ++t)
#pragma unroll
    for(int r=0;r<8;++r)
      pb[(r + 8*kh)*296 + t*16 + m15] = f2bf(sc[t][r] * inv[r]);
  __syncthreads();

  // ---- O = P Vp : 9 K-steps x 3 d-tiles
  v8f o0 = zero8(), o1 = zero8(), o2 = zero8();
  for(int ks = 0; ks < 288; ks += 32){
    v16bf ap = load_frag_a(&pb[m15*296 + ks], kh);
    v16bf b0 = load_frag_b(&vpL[( 0 + m15)*288 + ks], kh);
    v16bf b1 = load_frag_b(&vpL[(16 + m15)*288 + ks], kh);
    v16bf b2 = load_frag_b(&vpL[(32 + m15)*288 + ks], kh);
    o0 = WMMA_BF16(ap, b0, o0);
    o1 = WMMA_BF16(ap, b1, o1);
    o2 = WMMA_BF16(ap, b2, o2);
  }

  const int b_ = bh >> 3, h = bh & 7;
#pragma unroll
  for(int r=0;r<8;++r){
    size_t orow = (size_t)b_*4096 + rb*64 + wid*16 + r + 8*kh;
    size_t base = orow*384 + h*48 + m15;
    aobf[base     ] = f2bf(o0[r]);
    aobf[base + 16] = f2bf(o1[r]);
    aobf[base + 32] = f2bf(o2[r]);
  }
}

// ---------------------------------------------------------------- host
extern "C" void kernel_launch(void* const* d_in, const int* in_sizes, int n_in,
                              void* d_out, int out_size, void* d_ws, size_t ws_size,
                              hipStream_t stream)
{
  const float* x      = (const float*)d_in[0];
  const float* w_qkv  = (const float*)d_in[1];
  const float* w_proj = (const float*)d_in[2];
  const float* b_proj = (const float*)d_in[3];
  float* out = (float*)d_out;

  char* p = (char*)d_ws;
  auto take = [&](size_t bytes){ char* r = p; p += (bytes + 255) & ~(size_t)255; return r; };
  u16* x_bf     = (u16*)take((size_t)32768*384*2);
  u16* wqkv_bf  = (u16*)take((size_t)384*1152*2);
  u16* wproj_bf = (u16*)take((size_t)384*384*2);
  u16* q_bf     = (u16*)take((size_t)64*4096*64*2);   // hd padded to 64
  u16* k_bf     = (u16*)take((size_t)64*4096*48*2);
  u16* v_bf     = (u16*)take((size_t)64*4096*48*2);
  u16* kp_bf    = (u16*)take((size_t)64*288*64*2);
  u16* vpT_bf   = (u16*)take((size_t)64*48*288*2);
  u16* ao_bf    = (u16*)take((size_t)32768*384*2);

  cvt_f32_bf16<<<4096, 256, 0, stream>>>(x,      x_bf,     32768*384);
  cvt_f32_bf16<<<512,  256, 0, stream>>>(w_qkv,  wqkv_bf,  384*1152);
  cvt_f32_bf16<<<256,  256, 0, stream>>>(w_proj, wproj_bf, 384*384);
  zero_u16    <<<4096, 256, 0, stream>>>(q_bf, 64*4096*64);

  gemm_bf16_k384<<<dim3(256, 9), 256, 0, stream>>>(x_bf, wqkv_bf, 1152, 0,
      nullptr, nullptr, q_bf, k_bf, v_bf);

  psp_pool<<<64, 256, 0, stream>>>(k_bf, kp_bf, 0);
  psp_pool<<<64, 256, 0, stream>>>(v_bf, vpT_bf, 1);

  attn_psp<<<dim3(64, 64), 128, 0, stream>>>(q_bf, kp_bf, vpT_bf, ao_bf);

  gemm_bf16_k384<<<dim3(256, 3), 256, 0, stream>>>(ao_bf, wproj_bf, 384, 1,
      b_proj, out, nullptr, nullptr, nullptr);
}
